// MultiHeadAttention_44152263803528
// MI455X (gfx1250) — matrix operand order repaired and checked by reading, not compiled, not run on hardware
//
#include <hip/hip_runtime.h>
#include <hip/hip_bf16.h>

// ---------------------------------------------------------------------------
// GQA attention block for MI455X (gfx1250, wave32, WMMA bf16 + TDM)
//   B=1, S=2048, D=2048, G=4, HPG=8, H=32, DK=512, HD=64
// Pipeline: bf16-convert -> QKV GEMMs (WMMA, 32x64 wave tiles) -> rope ->
// attention (TDM double-buffered K/V tiles in LDS, WMMA, streaming
// exp/(sum+1) softmax) -> output GEMM (WMMA).
// ---------------------------------------------------------------------------

typedef __bf16 bf16_t;
typedef __attribute__((ext_vector_type(16))) __bf16 v16bf;
typedef __attribute__((ext_vector_type(8)))  __bf16 v8bf;
typedef __attribute__((ext_vector_type(8)))  float  v8f;
typedef __attribute__((ext_vector_type(4)))  unsigned int u32x4;
typedef __attribute__((ext_vector_type(8)))  int i32x8;
typedef __attribute__((ext_vector_type(4)))  int i32x4;

#define S_LEN  2048
#define D_DIM  2048
#define DK_DIM 512
#define H_HEADS 32

// A-fragment (16x32 bf16): one contiguous 16-element chunk per lane (two 16B loads).
__device__ __forceinline__ v16bf load_frag_a(const bf16_t* p0, const bf16_t* p1) {
  union { v16bf v; v8bf h[2]; } u;
  u.h[0] = *reinterpret_cast<const v8bf*>(p0);
  u.h[1] = *reinterpret_cast<const v8bf*>(p0 + 8); (void)p1;
  return u.v;
}

// B-fragment (32x16 bf16): one contiguous 16-element (32B) chunk per lane.
__device__ __forceinline__ v16bf load_frag_b(const bf16_t* p) {
  return *reinterpret_cast<const v16bf*>(p);
}

__device__ __forceinline__ v8f wmma_bf16(v16bf a, v16bf b, v8f c) {
  return __builtin_amdgcn_wmma_f32_16x16x32_bf16(false, a, false, b,
                                                 (short)0, c, false, false);
}

// TDM: 2-D tile (bf16 elements) Global -> LDS.  D# packed per ISA 08 §8:
//  g0: [1:0]=count=1, [63:32]=lds_addr, [120:64]=global_addr, [127:126]=2
//  g1: [17:16]=data_size(2B), dims/strides as below; g2/g3 zero (2-D tile).
// This toolchain exposes the 6-arg builtin (extra i32x8 group, then cpol).
__device__ __forceinline__ void tdm_load_2d_bf16(unsigned lds_off,
                                                 const bf16_t* gptr,
                                                 unsigned tensor_d0,
                                                 unsigned tensor_d1,
                                                 unsigned tile_d0,
                                                 unsigned tile_d1,
                                                 unsigned long d0_stride) {
  unsigned long ga = (unsigned long)gptr;
  u32x4 g0;
  g0[0] = 1u;                                   // count=1 (valid descriptor)
  g0[1] = lds_off;                              // LDS byte address
  g0[2] = (unsigned)(ga & 0xffffffffu);         // global_addr[31:0]
  g0[3] = (unsigned)(ga >> 32) | (2u << 30);    // global_addr[56:32] | type=2
  i32x8 g1;
  g1[0] = (int)(1u << 16);                      // data_size = 1 -> 2 bytes
  g1[1] = (int)((tensor_d0 & 0xffffu) << 16);                    // dim0[15:0]
  g1[2] = (int)((tensor_d0 >> 16) | ((tensor_d1 & 0xffffu) << 16));
  g1[3] = (int)((tensor_d1 >> 16) | (tile_d0 << 16));
  g1[4] = (int)(tile_d1 & 0xffffu);             // tile_dim1 (tile_dim2 = 0)
  g1[5] = (int)(d0_stride & 0xffffffffu);       // dim0_stride[31:0]
  g1[6] = (int)((d0_stride >> 32) & 0xffffu);   // dim0_stride[47:32]
  g1[7] = 0;
  i32x4 z4 = {0, 0, 0, 0};
  i32x8 z8 = {0, 0, 0, 0, 0, 0, 0, 0};
  __builtin_amdgcn_tensor_load_to_lds(g0, g1, z4, z4, z8, 0);
}

// ---------------------------------------------------------------------------
__global__ void cvt_f32_bf16(const float* __restrict__ in,
                             bf16_t* __restrict__ out, long n) {
  long i = (long)blockIdx.x * blockDim.x + threadIdx.x;
  if (i < n) out[i] = (bf16_t)in[i];
}

// rope: out[s,c] = rot(in)[s,c] * (pm0[s,c] + pm1[s,c]); rot = concat(-hi, lo)
__global__ void rope_bf16(const float* __restrict__ in,
                          const float* __restrict__ pm,   // [2,1,S,D]
                          bf16_t* __restrict__ out, int d, long n) {
  long i = (long)blockIdx.x * blockDim.x + threadIdx.x;
  if (i >= n) return;
  int s = (int)(i / d), c = (int)(i % d);
  int hd = d >> 1;
  float rx = (c < hd) ? -in[(long)s * d + c + hd] : in[(long)s * d + c - hd];
  long po = (long)s * D_DIM + c;
  float p = pm[po] + pm[(long)S_LEN * D_DIM + po];
  out[i] = (bf16_t)(rx * p);
}

// vT[c, s] = (bf16) v[s, c]   (v: [S, DK] fp32)
__global__ void transpose_bf16(const float* __restrict__ in,
                               bf16_t* __restrict__ out, long n) {
  long i = (long)blockIdx.x * blockDim.x + threadIdx.x;
  if (i >= n) return;
  int s = (int)(i / DK_DIM), c = (int)(i % DK_DIM);
  out[(long)c * S_LEN + s] = (bf16_t)in[i];
}

// ---------------------------------------------------------------------------
// C[M,N] (f32) = A[M,K] (bf16) @ W[N,K]^T (bf16). One wave -> 32x64 tile
// (2 A-fragments x 4 B-fragments -> 8 WMMAs per K-step; B reused 2x, A 4x).
__global__ void __launch_bounds__(128)
gemm_bf16_nt(const bf16_t* __restrict__ A, const bf16_t* __restrict__ W,
             float* __restrict__ C, int M, int N, int K) {
  const int lane  = threadIdx.x & 31;
  const int ln    = lane & 15;
  const int half  = lane >> 4;
  const int base0 = half * 16;       // A-fragment K split
  const int kb    = half * 16;       // B-fragment K split
  const int tileN = blockIdx.x;
  const int tileM = blockIdx.y * blockDim.y + threadIdx.y;   // 32-row tiles

  const long arow0 = (long)(tileM * 32 + ln) * K;
  const long arow1 = (long)(tileM * 32 + 16 + ln) * K;
  long wrow0 = (long)(tileN * 64 +  0 + ln) * K;
  long wrow1 = (long)(tileN * 64 + 16 + ln) * K;
  long wrow2 = (long)(tileN * 64 + 32 + ln) * K;
  long wrow3 = (long)(tileN * 64 + 48 + ln) * K;

  v8f acc[2][4] = {};
  for (int k = 0; k < K; k += 32) {
    __builtin_prefetch(A + arow0 + k + 256, 0, 3);
    __builtin_prefetch(W + wrow0 + k + 256, 0, 3);
    v16bf a0 = load_frag_a(A + arow0 + k + base0, A + arow0 + k + 16 + base0);
    v16bf a1 = load_frag_a(A + arow1 + k + base0, A + arow1 + k + 16 + base0);
    v16bf b0 = load_frag_b(W + wrow0 + k + kb);
    v16bf b1 = load_frag_b(W + wrow1 + k + kb);
    v16bf b2 = load_frag_b(W + wrow2 + k + kb);
    v16bf b3 = load_frag_b(W + wrow3 + k + kb);
    acc[0][0] = wmma_bf16(a0, b0, acc[0][0]);
    acc[0][1] = wmma_bf16(a0, b1, acc[0][1]);
    acc[0][2] = wmma_bf16(a0, b2, acc[0][2]);
    acc[0][3] = wmma_bf16(a0, b3, acc[0][3]);
    acc[1][0] = wmma_bf16(a1, b0, acc[1][0]);
    acc[1][1] = wmma_bf16(a1, b1, acc[1][1]);
    acc[1][2] = wmma_bf16(a1, b2, acc[1][2]);
    acc[1][3] = wmma_bf16(a1, b3, acc[1][3]);
  }
  const int col0 = tileN * 64 + ln;
#pragma unroll
  for (int mt = 0; mt < 2; ++mt) {
    const int row0 = tileM * 32 + mt * 16 + 8 * half;  // C layout: M = v+8*half
#pragma unroll
    for (int nt = 0; nt < 4; ++nt)
#pragma unroll
      for (int v = 0; v < 8; ++v)
        C[(long)(row0 + v) * N + col0 + nt * 16] = acc[mt][nt][v];
  }
}

// ---------------------------------------------------------------------------
// Attention. Block = 4 waves = 4 query tiles of ONE head. Wave 0 drives the
// Tensor Data Mover: per 32-key chunk it DMAs K (32x64) and V^T (64x32) tiles
// into double-buffered LDS, waits on TENSORcnt; the block syncs with split
// barriers; all 4 waves consume the tiles via ds_load fragments.
// Softmax variant faithful to reference: e/(sum(e)+1), +1e-10 above diagonal.
__global__ void __launch_bounds__(128)
attn_kernel(const bf16_t* __restrict__ qr,   // [S, D]  rope'd Q, bf16
            const bf16_t* __restrict__ kr,   // [S, DK] rope'd K, bf16
            const bf16_t* __restrict__ vT,   // [DK, S] V transposed, bf16
            bf16_t* __restrict__ outb) {     // [S, D]
  __shared__ bf16_t Kt[2][32][64];           // K tile:  rows=t, cols=k-dim
  __shared__ bf16_t Vt[2][64][32];           // V tile:  rows=head-dim, cols=t
  __shared__ bf16_t Pst[4][16][32];          // per-wave P staging

  const int lane  = threadIdx.x & 31;
  const int wid   = threadIdx.x >> 5;        // wave id: query tile select
  const int n     = lane & 15;
  const int half  = lane >> 4;
  const int base0 = half * 16;
  const int kb    = half * 16;
  const int stile = blockIdx.x * 4 + wid;    // query row tile
  const int h     = blockIdx.y;              // head
  const int qcol  = h * 64;
  const int kcol  = (h & 7) * 64;            // GQA broadcast slice

  // Q fragments (HD=64 -> K-dim 0..31 and 32..63), loaded once per wave.
  const long qoff = (long)(stile * 16 + n) * D_DIM + qcol;
  v16bf aq0 = load_frag_a(qr + qoff + base0,      qr + qoff + 16 + base0);
  v16bf aq1 = load_frag_a(qr + qoff + 32 + base0, qr + qoff + 48 + base0);

  v8f o0 = {}, o1 = {}, o2 = {}, o3 = {};
  v8f dp = {};
  const float scale = 1.0f / (22.6274170f + 1e-6f);   // 1/(sqrt(512)+eps)

  // Prime buffer 0.
  if (wid == 0) {
    tdm_load_2d_bf16((unsigned)(size_t)&Kt[0][0][0], kr + (long)0 * DK_DIM + kcol,
                     DK_DIM, S_LEN, 64, 32, DK_DIM);
    tdm_load_2d_bf16((unsigned)(size_t)&Vt[0][0][0], vT + (long)kcol * S_LEN + 0,
                     S_LEN, DK_DIM, 32, 64, S_LEN);
  }

  for (int t0 = 0; t0 < S_LEN; t0 += 32) {
    const int buf = (t0 >> 5) & 1;
    if (wid == 0) {
      if (t0 + 32 < S_LEN) {       // DMA next chunk into the other buffer
        const int tn = t0 + 32, nb = buf ^ 1;
        tdm_load_2d_bf16((unsigned)(size_t)&Kt[nb][0][0],
                         kr + (long)tn * DK_DIM + kcol,
                         DK_DIM, S_LEN, 64, 32, DK_DIM);
        tdm_load_2d_bf16((unsigned)(size_t)&Vt[nb][0][0],
                         vT + (long)kcol * S_LEN + tn,
                         S_LEN, DK_DIM, 32, 64, S_LEN);
        __builtin_amdgcn_s_wait_tensorcnt(2);   // current chunk's pair done
      } else {
        __builtin_amdgcn_s_wait_tensorcnt(0);
      }
    }
    __syncthreads();                            // tiles visible to all waves
    asm volatile("" ::: "memory");

    // scores: Q(16x64) @ K^T(64x32) as two 16x16 subtiles, K from LDS.
    v16bf bA0 = load_frag_b(&Kt[buf][n][kb]);
    v16bf bA1 = load_frag_b(&Kt[buf][n][32 + kb]);
    v16bf bB0 = load_frag_b(&Kt[buf][16 + n][kb]);
    v16bf bB1 = load_frag_b(&Kt[buf][16 + n][32 + kb]);

    v8f s0 = {}, s1 = {};
    s0 = wmma_bf16(aq0, bA0, s0);
    s0 = wmma_bf16(aq1, bA1, s0);
    s1 = wmma_bf16(aq0, bB0, s1);
    s1 = wmma_bf16(aq1, bB1, s1);

    // exp(score*scale + mask), denominator partials, stage P into LDS.
#pragma unroll
    for (int v = 0; v < 8; ++v) {
      const int srow = stile * 16 + v + 8 * half;
      const int tA = t0 + n, tB = t0 + 16 + n;
      float e0 = __expf(s0[v] * scale + ((tA > srow) ? 1e-10f : 0.0f));
      float e1 = __expf(s1[v] * scale + ((tB > srow) ? 1e-10f : 0.0f));
      dp[v] += e0 + e1;
      Pst[wid][v + 8 * half][n]      = (bf16_t)e0;
      Pst[wid][v + 8 * half][16 + n] = (bf16_t)e1;
    }

    // Re-fragment P (16x32) as A-operand; V fragments from LDS.
    v16bf pf = load_frag_a(&Pst[wid][n][base0], &Pst[wid][n][16 + base0]);
    o0 = wmma_bf16(pf, load_frag_b(&Vt[buf][ 0 + n][kb]), o0);
    o1 = wmma_bf16(pf, load_frag_b(&Vt[buf][16 + n][kb]), o1);
    o2 = wmma_bf16(pf, load_frag_b(&Vt[buf][32 + n][kb]), o2);
    o3 = wmma_bf16(pf, load_frag_b(&Vt[buf][48 + n][kb]), o3);

    __syncthreads();          // everyone done with buf before it is re-filled
  }

  // Row-sum across each 16-lane half; lands on the (lane,vgpr) owning the row.
#pragma unroll
  for (int v = 0; v < 8; ++v) {
    float d = dp[v];
    d += __shfl_xor(d, 1);
    d += __shfl_xor(d, 2);
    d += __shfl_xor(d, 4);
    d += __shfl_xor(d, 8);
    dp[v] = d + 1.0f;         // reference: sum(e) + 1
  }

  v8f os[4] = {o0, o1, o2, o3};
#pragma unroll
  for (int nt = 0; nt < 4; ++nt)
#pragma unroll
    for (int v = 0; v < 8; ++v) {
      const long r = stile * 16 + v + 8 * half;
      outb[r * D_DIM + qcol + nt * 16 + n] = (bf16_t)(os[nt][v] / dp[v]);
    }
}

// ---------------------------------------------------------------------------
extern "C" void kernel_launch(void* const* d_in, const int* in_sizes, int n_in,
                              void* d_out, int out_size, void* d_ws, size_t ws_size,
                              hipStream_t stream) {
  (void)in_sizes; (void)n_in; (void)out_size; (void)ws_size;
  const float* x  = (const float*)d_in[0];
  const float* pm = (const float*)d_in[1];
  const float* Wq = (const float*)d_in[2];
  const float* Wk = (const float*)d_in[3];
  const float* Wv = (const float*)d_in[4];
  const float* Wo = (const float*)d_in[5];
  float* out = (float*)d_out;

  const long nSD = (long)S_LEN * D_DIM;     // 4M
  const long nDD = (long)D_DIM * D_DIM;     // 4M
  const long nKD = (long)DK_DIM * D_DIM;    // 1M
  const long nSK = (long)S_LEN * DK_DIM;    // 1M

  char* w = (char*)d_ws;
  auto carve = [&](size_t bytes) {
    char* p = w;
    w += (bytes + 255) & ~(size_t)255;
    return p;
  };
  bf16_t* xb    = (bf16_t*)carve(nSD * 2);
  bf16_t* wqb   = (bf16_t*)carve(nDD * 2);
  bf16_t* wkb   = (bf16_t*)carve(nKD * 2);
  bf16_t* wvb   = (bf16_t*)carve(nKD * 2);
  bf16_t* wob   = (bf16_t*)carve(nDD * 2);
  float*  xq    = (float*) carve(nSD * 4);
  float*  xk    = (float*) carve(nSK * 4);
  float*  xv    = (float*) carve(nSK * 4);
  bf16_t* qrb   = (bf16_t*)carve(nSD * 2);
  bf16_t* krb   = (bf16_t*)carve(nSK * 2);
  bf16_t* vTb   = (bf16_t*)carve(nSK * 2);
  bf16_t* attnb = (bf16_t*)carve(nSD * 2);

  const int T = 256;
  auto blks = [&](long n) { return (unsigned)((n + T - 1) / T); };

  // 1) bf16 conversions
  cvt_f32_bf16<<<blks(nSD), T, 0, stream>>>(x,  xb,  nSD);
  cvt_f32_bf16<<<blks(nDD), T, 0, stream>>>(Wq, wqb, nDD);
  cvt_f32_bf16<<<blks(nKD), T, 0, stream>>>(Wk, wkb, nKD);
  cvt_f32_bf16<<<blks(nKD), T, 0, stream>>>(Wv, wvb, nKD);
  cvt_f32_bf16<<<blks(nDD), T, 0, stream>>>(Wo, wob, nDD);

  // 2) projections: xq = x Wq^T, xk = x Wk^T, xv = x Wv^T
  dim3 gblk(32, 4);
  gemm_bf16_nt<<<dim3(D_DIM / 64,  S_LEN / 128), gblk, 0, stream>>>(xb, wqb, xq, S_LEN, D_DIM,  D_DIM);
  gemm_bf16_nt<<<dim3(DK_DIM / 64, S_LEN / 128), gblk, 0, stream>>>(xb, wkb, xk, S_LEN, DK_DIM, D_DIM);
  gemm_bf16_nt<<<dim3(DK_DIM / 64, S_LEN / 128), gblk, 0, stream>>>(xb, wvb, xv, S_LEN, DK_DIM, D_DIM);

  // 3) position embedding (quirky rope) -> bf16; transpose V -> bf16
  rope_bf16<<<blks(nSD), T, 0, stream>>>(xq, pm, qrb, D_DIM,  nSD);
  rope_bf16<<<blks(nSK), T, 0, stream>>>(xk, pm, krb, DK_DIM, nSK);
  transpose_bf16<<<blks(nSK), T, 0, stream>>>(xv, vTb, nSK);

  // 4) attention: 32 heads x 32 blocks x (4 waves = 4 query tiles)
  attn_kernel<<<dim3(S_LEN / 64, H_HEADS), 128, 0, stream>>>(qrb, krb, vTb, attnb);

  // 5) output projection: out = attn Wo^T (f32 result)
  gemm_bf16_nt<<<dim3(D_DIM / 64, S_LEN / 128), gblk, 0, stream>>>(attnb, wob, out, S_LEN, D_DIM, D_DIM);
}
